// Policy_Network_2164663517985
// MI455X (gfx1250) — matrix operand order repaired and checked by reading, not compiled, not run on hardware
//
#include <hip/hip_runtime.h>
#include <hip/hip_bf16.h>
#include <stdint.h>

typedef __attribute__((ext_vector_type(16))) __bf16 v16bf;
typedef __attribute__((ext_vector_type(8)))  __bf16 v8bf;
typedef __attribute__((ext_vector_type(8)))  float  v8f;
typedef int v4i_vs __attribute__((__vector_size__(16)));

#define AS1 __attribute__((address_space(1)))
#define AS3 __attribute__((address_space(3)))

#if defined(__has_builtin)
#if __has_builtin(__builtin_amdgcn_global_load_async_to_lds_b128)
#define HAVE_ASYNC_LDS 1
#endif
#endif
#ifndef HAVE_ASYNC_LDS
#define HAVE_ASYNC_LDS 0
#endif

#define B_ROWS 8192
#define K_CTX  5152   // INPUT=5124 padded up to a multiple of 32

// ---------------------------------------------------------------------------
// Build ctx = concat(state, task, action) as bf16, K padded with zeros.
// ---------------------------------------------------------------------------
__global__ __launch_bounds__(256) void build_ctx(
    const float* __restrict__ state, const float* __restrict__ action,
    const float* __restrict__ task, __bf16* __restrict__ ctx)
{
  size_t idx = (size_t)blockIdx.x * 256 + threadIdx.x;
  size_t total = (size_t)B_ROWS * K_CTX;
  if (idx >= total) return;
  int col = (int)(idx % K_CTX);
  int row = (int)(idx / K_CTX);
  float v = 0.0f;
  if (col < 4096)      v = state [(size_t)row * 4096 + col];
  else if (col < 4100) v = task  [(size_t)row * 4    + (col - 4096)];
  else if (col < 5124) v = action[(size_t)row * 1024 + (col - 4100)];
  ctx[idx] = (__bf16)v;
}

// ---------------------------------------------------------------------------
// Transpose-convert W[K,N] f32 (row-major, row stride N) -> Wt[N,Kpad] bf16,
// zero padding K..Kpad-1.  LDS 32x32 tile so both sides stay coalesced.
// ---------------------------------------------------------------------------
__global__ __launch_bounds__(256) void conv_w(
    const float* __restrict__ W, __bf16* __restrict__ Wt,
    int K, int N, int Kpad)
{
  __shared__ float tile[32][33];
  const int tx = threadIdx.x & 31;
  const int ty = threadIdx.x >> 5;      // 0..7
  const int kb = blockIdx.y * 32;
  const int nb = blockIdx.x * 32;
#pragma unroll
  for (int i = 0; i < 4; ++i) {
    int r = ty + i * 8;
    int k = kb + r;
    tile[r][tx] = (k < K) ? W[(size_t)k * N + nb + tx] : 0.0f;
  }
  __syncthreads();
#pragma unroll
  for (int i = 0; i < 4; ++i) {
    int r = ty + i * 8;
    Wt[(size_t)(nb + r) * Kpad + kb + tx] = (__bf16)tile[tx][r];
  }
}

// ---------------------------------------------------------------------------
// 16B async copy global -> LDS (no VGPR staging, tracked by ASYNCcnt).
// ---------------------------------------------------------------------------
__device__ __forceinline__ void async_ld16(const __bf16* g, __bf16* l) {
#if HAVE_ASYNC_LDS
  __builtin_amdgcn_global_load_async_to_lds_b128(
      (AS1 v4i_vs*)(void*)g, (AS3 v4i_vs*)(void*)l, 0, 0);
#else
  (void)g; (void)l;
#endif
}
__device__ __forceinline__ void wait_async0() {
#if HAVE_ASYNC_LDS
  asm volatile("s_wait_asynccnt 0x0" ::: "memory");
#endif
}

// ---------------------------------------------------------------------------
// GEMM: D[M,N] = A[M,K]_bf16 * Bt[N,K]_bf16^T + (ADD_C ? Cadd[M,N] : bias[N])
// Block tile 128x256x32, 8 waves, each wave = 4x4 WMMA 16x16x32 bf16 tiles.
// Double-buffered LDS; tile s+1 streams HBM->LDS via async loads while the
// wave runs 16 WMMAs on tile s; one s_wait_asynccnt + barrier per K-step.
// ---------------------------------------------------------------------------
template <bool OUT_BF16, bool ADD_C>
__global__ __launch_bounds__(256) void gemm_wmma_bf16(
    const __bf16* __restrict__ A,      // [M, K]
    const __bf16* __restrict__ Bt,     // [N, K]  (pre-transposed weights)
    const float*  __restrict__ addsrc, // bias[N] (!ADD_C) or Cadd[M,N] (ADD_C)
    void* __restrict__ Dv, int N, int K)
{
  constexpr int BM = 128, BN = 256, BK = 32, LDSS = BK + 8; // row = 40 bf16 = 80B
  __shared__ __bf16 As[2][BM * LDSS];
  __shared__ __bf16 Bs[2][BN * LDSS];

  const int tid   = threadIdx.x;
  const int lane  = tid & 31;
  const int wave  = tid >> 5;
  const int waveM = wave >> 2;       // 0..1  (64-row slab)
  const int waveN = wave & 3;        // 0..3  (64-col slab)
  const int hi    = lane >> 4;       // lane half selects K sub-chunks
  const int ml    = lane & 15;

  const int m0 = blockIdx.y * BM;
  const int n0 = blockIdx.x * BN;

  // global->LDS distribution: each thread owns one 8-bf16 chunk per 64 rows
  const int lrow = tid >> 2;           // 0..63
  const int lchk = (tid & 3) * 8;      // element offset within 32-wide row
  const __bf16* Abase = A  + (size_t)(m0 + lrow) * K + lchk;
  const __bf16* Bbase = Bt + (size_t)(n0 + lrow) * K + lchk;
  const int ldsoff = lrow * LDSS + lchk;

  v8f acc[4][4] = {};

#if HAVE_ASYNC_LDS
  auto fetch_tile = [&](int s, int buf) {
    const int kt = s * BK;
#pragma unroll
    for (int it = 0; it < 2; ++it)
      async_ld16(Abase + (size_t)it * 64 * K + kt, &As[buf][ldsoff + it * 64 * LDSS]);
#pragma unroll
    for (int it = 0; it < 4; ++it)
      async_ld16(Bbase + (size_t)it * 64 * K + kt, &Bs[buf][ldsoff + it * 64 * LDSS]);
  };
#else
  auto fetch_tile = [&](int s, int buf) {
    const int kt = s * BK;
#pragma unroll
    for (int it = 0; it < 2; ++it)
      *reinterpret_cast<uint4*>(&As[buf][ldsoff + it * 64 * LDSS]) =
          *reinterpret_cast<const uint4*>(Abase + (size_t)it * 64 * K + kt);
#pragma unroll
    for (int it = 0; it < 4; ++it)
      *reinterpret_cast<uint4*>(&Bs[buf][ldsoff + it * 64 * LDSS]) =
          *reinterpret_cast<const uint4*>(Bbase + (size_t)it * 64 * K + kt);
  };
#endif

  const int nk = K / BK;
  fetch_tile(0, 0);
  wait_async0();
  __syncthreads();

  for (int s = 0; s < nk; ++s) {
    const int cur = s & 1;
    if (s + 1 < nk) {
      fetch_tile(s + 1, cur ^ 1);      // streams into the idle buffer
      if (s + 2 < nk) {
        __builtin_prefetch(Abase + (size_t)(s + 2) * BK);
        __builtin_prefetch(Bbase + (size_t)(s + 2) * BK);
      }
    }

    const __bf16* as = As[cur];
    const __bf16* bs = Bs[cur];

    // A (16x32): lane<16 holds K[0:8]+K[16:24]; lane>=16 holds K[8:16]+K[24:32]
    v16bf af[4];
#pragma unroll
    for (int mi = 0; mi < 4; ++mi) {
      int m = waveM * 64 + mi * 16 + ml;
      v8bf a0 = *reinterpret_cast<const v8bf*>(&as[m * LDSS + hi * 8]);
      v8bf a1 = *reinterpret_cast<const v8bf*>(&as[m * LDSS + 16 + hi * 8]);
      af[mi] = __builtin_shufflevector(a0, a1, 0,1,2,3,4,5,6,7,8,9,10,11,12,13,14,15);
    }
    // B (32x16): lane<16 holds K[0:8]+K[16:24] of col ml; lane>=16 holds K[8:16]+K[24:32]
    v16bf bfr[4];
#pragma unroll
    for (int ni = 0; ni < 4; ++ni) {
      int n = waveN * 64 + ni * 16 + ml;
      v8bf b0 = *reinterpret_cast<const v8bf*>(&bs[n * LDSS + hi * 8]);
      v8bf b1 = *reinterpret_cast<const v8bf*>(&bs[n * LDSS + 16 + hi * 8]);
      bfr[ni] = __builtin_shufflevector(b0, b1, 0,1,2,3,4,5,6,7,8,9,10,11,12,13,14,15);
    }

#pragma unroll
    for (int mi = 0; mi < 4; ++mi)
#pragma unroll
      for (int ni = 0; ni < 4; ++ni)
        acc[mi][ni] = __builtin_amdgcn_wmma_f32_16x16x32_bf16(
            false, af[mi], false, bfr[ni], (short)0, acc[mi][ni], false, false);

    if (s + 1 < nk) {
      wait_async0();                   // this wave's async tile is in LDS
      __syncthreads();                 // everyone's tile is in LDS
    }
  }

  // ---- epilogue: C/D layout: VGPR r -> row (hi?8:0)+r, col = ml ----
#pragma unroll
  for (int mi = 0; mi < 4; ++mi) {
#pragma unroll
    for (int ni = 0; ni < 4; ++ni) {
      const int row0 = m0 + waveM * 64 + mi * 16 + hi * 8;
      const int col  = n0 + waveN * 64 + ni * 16 + ml;
      float bv = 0.0f;
      if (!ADD_C) bv = addsrc[col];
      size_t off = (size_t)row0 * N + col;
#pragma unroll
      for (int r = 0; r < 8; ++r, off += (size_t)N) {
        float v = acc[mi][ni][r] + (ADD_C ? addsrc[off] : bv);
        if (OUT_BF16) reinterpret_cast<__bf16*>(Dv)[off] = (__bf16)v;
        else          reinterpret_cast<float*>(Dv)[off]  = v;
      }
    }
  }
}

// ---------------------------------------------------------------------------
extern "C" void kernel_launch(void* const* d_in, const int* in_sizes, int n_in,
                              void* d_out, int out_size, void* d_ws, size_t ws_size,
                              hipStream_t stream)
{
  const float* state  = (const float*)d_in[0];
  const float* action = (const float*)d_in[1];
  const float* task   = (const float*)d_in[2];
  // cx* weights (indices 3..18) are dead: _kwta is the identity map.
  const float* W_l1 = (const float*)d_in[19];
  const float* b_l1 = (const float*)d_in[20];
  const float* W_l2 = (const float*)d_in[21];
  const float* b_l2 = (const float*)d_in[22];
  const float* W_l3 = (const float*)d_in[23];
  const float* b_l3 = (const float*)d_in[24];
  const float* W_l4 = (const float*)d_in[25];
  const float* b_l4 = (const float*)d_in[26];
  const float* W_l5 = (const float*)d_in[27];
  const float* b_l5 = (const float*)d_in[28];

  char* ws = (char*)d_ws;
  size_t off = 0;
  auto alloc = [&](size_t bytes) -> char* {
    char* p = ws + off;
    off += (bytes + 255) & ~(size_t)255;
    return p;
  };

  __bf16* ctx  = (__bf16*)alloc((size_t)B_ROWS * K_CTX * 2);
  __bf16* W1t  = (__bf16*)alloc((size_t)2048 * K_CTX * 2);
  __bf16* W2at = (__bf16*)alloc((size_t)1024 * 2048 * 2);
  __bf16* W2bt = (__bf16*)alloc((size_t)1024 * K_CTX * 2);
  __bf16* W3at = (__bf16*)alloc((size_t)1024 * 1024 * 2);
  __bf16* W3bt = (__bf16*)alloc((size_t)1024 * K_CTX * 2);
  __bf16* W4at = (__bf16*)alloc((size_t)512 * 1024 * 2);
  __bf16* W4bt = (__bf16*)alloc((size_t)512 * K_CTX * 2);
  __bf16* W5t  = (__bf16*)alloc((size_t)1024 * 512 * 2);
  __bf16* x1   = (__bf16*)alloc((size_t)B_ROWS * 2048 * 2);
  __bf16* x2   = (__bf16*)alloc((size_t)B_ROWS * 1024 * 2);
  __bf16* x3   = (__bf16*)alloc((size_t)B_ROWS * 1024 * 2);
  __bf16* x4   = (__bf16*)alloc((size_t)B_ROWS * 512 * 2);
  float*  p2   = (float*) alloc((size_t)B_ROWS * 1024 * 4);
  float*  p3   = (float*) alloc((size_t)B_ROWS * 1024 * 4);
  float*  p4   = (float*) alloc((size_t)B_ROWS * 512 * 4);

  // 1) ctx (concat + bf16 + pad)
  {
    size_t total = (size_t)B_ROWS * K_CTX;
    build_ctx<<<(unsigned)((total + 255) / 256), 256, 0, stream>>>(state, action, task, ctx);
  }

  // 2) weight transpose-convert (top = x-part, bottom = ctx-part of concat)
  conv_w<<<dim3(2048/32, K_CTX/32), 256, 0, stream>>>(W_l1,                        W1t, 5124, 2048, K_CTX);
  conv_w<<<dim3(1024/32, 2048/32),  256, 0, stream>>>(W_l2,                        W2at, 2048, 1024, 2048);
  conv_w<<<dim3(1024/32, K_CTX/32), 256, 0, stream>>>(W_l2 + (size_t)2048 * 1024,  W2bt, 5124, 1024, K_CTX);
  conv_w<<<dim3(1024/32, 1024/32),  256, 0, stream>>>(W_l3,                        W3at, 1024, 1024, 1024);
  conv_w<<<dim3(1024/32, K_CTX/32), 256, 0, stream>>>(W_l3 + (size_t)1024 * 1024,  W3bt, 5124, 1024, K_CTX);
  conv_w<<<dim3( 512/32, 1024/32),  256, 0, stream>>>(W_l4,                        W4at, 1024,  512, 1024);
  conv_w<<<dim3( 512/32, K_CTX/32), 256, 0, stream>>>(W_l4 + (size_t)1024 * 512,   W4bt, 5124,  512, K_CTX);
  conv_w<<<dim3(1024/32,  512/32),  256, 0, stream>>>(W_l5,                        W5t,   512, 1024,  512);

  const dim3 blk(256);
  const int MB = B_ROWS / 128; // 64

  // 3) ctx-side partials (f32, bias folded in)
  gemm_wmma_bf16<false,false><<<dim3(1024/256, MB), blk, 0, stream>>>(ctx, W2bt, b_l2, p2, 1024, K_CTX);
  gemm_wmma_bf16<false,false><<<dim3(1024/256, MB), blk, 0, stream>>>(ctx, W3bt, b_l3, p3, 1024, K_CTX);
  gemm_wmma_bf16<false,false><<<dim3( 512/256, MB), blk, 0, stream>>>(ctx, W4bt, b_l4, p4,  512, K_CTX);

  // 4) main chain
  gemm_wmma_bf16<true ,false><<<dim3(2048/256, MB), blk, 0, stream>>>(ctx, W1t,  b_l1, x1, 2048, K_CTX);
  gemm_wmma_bf16<true ,true ><<<dim3(1024/256, MB), blk, 0, stream>>>(x1,  W2at, p2,   x2, 1024, 2048);
  gemm_wmma_bf16<true ,true ><<<dim3(1024/256, MB), blk, 0, stream>>>(x2,  W3at, p3,   x3, 1024, 1024);
  gemm_wmma_bf16<true ,true ><<<dim3( 512/256, MB), blk, 0, stream>>>(x3,  W4at, p4,   x4,  512, 1024);

  // 5) output (f32)
  gemm_wmma_bf16<false,false><<<dim3(1024/256, MB), blk, 0, stream>>>(x4,  W5t,  b_l5, (float*)d_out, 1024, 512);

  (void)in_sizes; (void)n_in; (void)out_size; (void)ws_size;
}
